// DualDirectionMambaBlock_11544872092167
// MI455X (gfx1250) — compile-verified, matrix operand order checked by reading, not run on hardware
//
#include <hip/hip_runtime.h>

// Problem dims (compile-time)
#define B_   8
#define L_   2048
#define D_   256
#define DS_  16
#define DC_  4
#define DI_  512
#define DTR_ 16
#define DFF_ 1024
#define NROWS (B_ * L_)   // 16384

typedef __bf16 bf16_t;
typedef __attribute__((ext_vector_type(8)))  __bf16 v8bf;
typedef __attribute__((ext_vector_type(16))) __bf16 v16bf;
typedef __attribute__((ext_vector_type(8)))  float  v8f;
typedef __attribute__((ext_vector_type(4)))  int    v4i;

#if defined(__AMDGCN__) && __has_builtin(__builtin_amdgcn_global_load_async_to_lds_b128)
#define HAVE_ASYNC_LDS 1
#else
#define HAVE_ASYNC_LDS 0
#endif

__device__ __forceinline__ bf16_t to_bf16(float f) {
  unsigned u = __builtin_bit_cast(unsigned, f);
  unsigned r = u + 0x7fffu + ((u >> 16) & 1u);   // round-to-nearest-even
  unsigned short h = (unsigned short)(r >> 16);
  return __builtin_bit_cast(bf16_t, h);
}

// ---------------------------------------------------------------------------
// Weight convert + transpose: src[R x C] f32 row-major -> dst[C x R] bf16
// ---------------------------------------------------------------------------
__global__ void k_transpose_bf16(const float* __restrict__ src, bf16_t* __restrict__ dst,
                                 int R, int C) {
  long i = (long)blockIdx.x * 256 + threadIdx.x;
  if (i >= (long)R * C) return;
  int r = (int)(i / C), c = (int)(i % C);
  dst[(long)c * R + r] = to_bf16(src[i]);
}

__global__ void k_copy_f32(const float* __restrict__ src, float* __restrict__ dst, long n) {
  long i = (long)blockIdx.x * 256 + threadIdx.x;
  if (i < n) dst[i] = src[i];
}

// ---------------------------------------------------------------------------
// LayerNorm over D=256, one block (256 thr) per row, bf16 out (GEMM A operand)
// ---------------------------------------------------------------------------
__global__ void k_layernorm_bf16(const float* __restrict__ x, const float* __restrict__ g,
                                 const float* __restrict__ b, bf16_t* __restrict__ out) {
  int row = blockIdx.x;
  int t = threadIdx.x;
  float v = x[(long)row * D_ + t];
  float s = v, s2 = v * v;
  #pragma unroll
  for (int o = 16; o; o >>= 1) { s += __shfl_xor(s, o, 32); s2 += __shfl_xor(s2, o, 32); }
  __shared__ float red[16];
  int wid = t >> 5;
  if ((t & 31) == 0) { red[wid] = s; red[wid + 8] = s2; }
  __syncthreads();
  float ts = 0.f, ts2 = 0.f;
  #pragma unroll
  for (int i = 0; i < 8; ++i) { ts += red[i]; ts2 += red[i + 8]; }
  float mu  = ts * (1.f / D_);
  float var = ts2 * (1.f / D_) - mu * mu;
  float inv = rsqrtf(var + 1e-5f);
  out[(long)row * D_ + t] = to_bf16((v - mu) * inv * g[t] + b[t]);
}

// ---------------------------------------------------------------------------
// WMMA bf16 GEMM: D[M,N] = A[M,K] @ B[K,N] (+bias), B supplied transposed
// (BT is [N,K] row-major bf16).
// Block = 256 thr = 8 waves along M (M-tile 128). Each wave owns a 16 x
// (16*NT) strip: one A fragment per K-step reused across NT WMMAs.
// B tile (NT*16 rows x 32 K) staged once per block into double-buffered LDS
// via the CDNA5 async-to-LDS path (ASYNCcnt) so stage(k+1) overlaps wmma(k).
// ---------------------------------------------------------------------------
__device__ __forceinline__ v16bf load_frag16x32(const bf16_t* __restrict__ base, int ld) {
  // CDNA5 16-bit 16x32 fragment: lanes 0-15 hold K 0..7 / 16..23,
  // lanes 16-31 hold K 8..15 / 24..31 (row M = lane & 15).
  int lane = threadIdx.x & 31;
  int r  = lane & 15;
  int hi = lane >> 4;
  const bf16_t* p = base + (long)r * ld + hi * 8;
  v8bf lo = *(const v8bf*)(p);
  v8bf hh = *(const v8bf*)(p + 16);
  return __builtin_shufflevector(lo, hh, 0,1,2,3,4,5,6,7,8,9,10,11,12,13,14,15);
}

enum { EPI_F32 = 0, EPI_GELU_BF16 = 1, EPI_ADD_F32 = 2 };

template <int EPI, int NT>
__global__ void k_gemm_wmma(const bf16_t* __restrict__ A, const bf16_t* __restrict__ BT,
                            const float* __restrict__ bias,
                            float* __restrict__ outF, bf16_t* __restrict__ outH,
                            int M, int N, int K, int ldo) {
  __shared__ bf16_t Bs[2][NT * 16 * 32];
  int t    = threadIdx.x;
  int wave = t >> 5;                        // 0..7 along M
  int m0 = blockIdx.y * 128 + wave * 16;
  int n0 = blockIdx.x * (16 * NT);
  const bf16_t* Ab = A + (long)m0 * K;

  // Stage BT rows [n0, n0+NT*16) x 32 K values into Bs[buf] (16B per thread-chunk).
  auto stage = [&](int buf, int kb) {
    for (int i = t; i < NT * 16 * 4; i += 256) {
      int row = i >> 2;
      int c   = (i & 3) * 8;
      const bf16_t* src = BT + (long)(n0 + row) * K + kb + c;
      bf16_t* dst = &Bs[buf][row * 32 + c];
#if HAVE_ASYNC_LDS
      __builtin_amdgcn_global_load_async_to_lds_b128(
          (__attribute__((address_space(1))) v4i*)src,
          (__attribute__((address_space(3))) v4i*)dst, 0, 0);
#else
      *(v8bf*)dst = *(const v8bf*)src;
#endif
    }
  };

  v8f acc[NT];
  #pragma unroll
  for (int j = 0; j < NT; ++j) acc[j] = (v8f){};

  const int nk = K / 32;
  stage(0, 0);
  for (int ki = 0; ki < nk; ++ki) {
    int buf = ki & 1;
#if HAVE_ASYNC_LDS
#if __has_builtin(__builtin_amdgcn_s_wait_asynccnt)
    __builtin_amdgcn_s_wait_asynccnt(0);
#else
    asm volatile("s_wait_asynccnt 0x0" ::: "memory");
#endif
#endif
    __syncthreads();                         // Bs[buf] ready; prior reads of buf^1 done
    if (ki + 1 < nk) stage(buf ^ 1, (ki + 1) * 32);
    __builtin_prefetch((const void*)(Ab + (ki + 2) * 32), 0, 1);  // global_prefetch_b8
    v16bf a = load_frag16x32(Ab + ki * 32, K);
    #pragma unroll
    for (int j = 0; j < NT; ++j) {
      v16bf b = load_frag16x32(&Bs[buf][j * 16 * 32], 32);
      acc[j] = __builtin_amdgcn_wmma_f32_16x16x32_bf16(false, a, false, b,
                                                       (short)0, acc[j], false, false);
    }
  }

  int lane = threadIdx.x & 31;
  int r    = lane & 15;                     // D: N = lane & 15
  int hi   = lane >> 4;                     // D: M = v + 8*hi
  #pragma unroll
  for (int j = 0; j < NT; ++j) {
    int col = n0 + j * 16 + r;
    float bv = bias ? bias[col] : 0.f;
    #pragma unroll
    for (int v = 0; v < 8; ++v) {
      int row = m0 + v + 8 * hi;
      float val = acc[j][v] + bv;
      if (EPI == EPI_F32) {
        outF[(long)row * ldo + col] = val;
      } else if (EPI == EPI_GELU_BF16) {
        float c3 = val + 0.044715f * val * val * val;
        float ge = 0.5f * val * (1.f + tanhf(0.7978845608028654f * c3));
        outH[(long)row * ldo + col] = to_bf16(ge);
      } else { // EPI_ADD_F32 (residual accumulate)
        outF[(long)row * ldo + col] += val;
      }
    }
  }
}

// ---------------------------------------------------------------------------
// Depthwise causal conv (width 4) + SiLU. xp = xz[:, :DI]. Dual f32/bf16 out.
// ---------------------------------------------------------------------------
__global__ void k_conv_silu(const float* __restrict__ xz, const float* __restrict__ cw,
                            const float* __restrict__ cb,
                            float* __restrict__ xcF, bf16_t* __restrict__ xcH) {
  long i = (long)blockIdx.x * 256 + threadIdx.x;    // over NROWS*DI
  if (i >= (long)NROWS * DI_) return;
  int  c = (int)(i % DI_);
  long n = i / DI_;
  int  l = (int)(n % L_);
  float acc = cb[c];
  #pragma unroll
  for (int j = 0; j < DC_; ++j) {
    int ll = l - (DC_ - 1) + j;
    if (ll >= 0) acc += cw[j * DI_ + c] * xz[(n - (DC_ - 1) + j) * (2 * DI_) + c];
  }
  float s = acc / (1.f + __expf(-acc));             // SiLU
  xcF[n * DI_ + c] = s;
  xcH[n * DI_ + c] = to_bf16(s);
}

// ---------------------------------------------------------------------------
// dt = softplus(dtr @ Wdt + bdt); dtr = dbl[:, :16]. K=16 -> VALU kernel.
// grid (NROWS, 2), block 256.
// ---------------------------------------------------------------------------
__global__ void k_dt_softplus(const float* __restrict__ dbl, const float* __restrict__ Wdt,
                              const float* __restrict__ bdt, float* __restrict__ dt) {
  int n = blockIdx.x;
  int d = blockIdx.y * 256 + threadIdx.x;
  __shared__ float dtr[DTR_];
  if (threadIdx.x < DTR_) dtr[threadIdx.x] = dbl[(long)n * 48 + threadIdx.x];
  __syncthreads();
  float acc = bdt[d];
  #pragma unroll
  for (int t = 0; t < DTR_; ++t) acc += dtr[t] * Wdt[t * DI_ + d];
  float sp = (acc > 20.f) ? acc : logf(1.f + __expf(acc));
  dt[(long)n * DI_ + d] = sp;
}

// ---------------------------------------------------------------------------
// Selective scan + Dp skip + SiLU(z) gating -> bf16 (A operand of Wout GEMM).
// grid (B, DI/16), block 256 = 16 d x 16 s. Each thread owns state h[d,s];
// sum over s via 16-lane shfl_xor butterflies inside the wave32.
// ---------------------------------------------------------------------------
__global__ void k_scan(const float* __restrict__ dt, const float* __restrict__ dbl,
                       const float* __restrict__ xcF, const float* __restrict__ Alog,
                       const float* __restrict__ Dp, const float* __restrict__ xz,
                       bf16_t* __restrict__ ygate) {
  int b    = blockIdx.x;
  int dblk = blockIdx.y;
  int t = threadIdx.x;
  int s  = t & 15;
  int dd = t >> 4;
  int d = dblk * 16 + dd;
  float Ads = -__expf(Alog[d * DS_ + s]);
  float Dpd = Dp[d];
  float h = 0.f;
  for (int l = 0; l < L_; ++l) {
    long n = (long)b * L_ + l;
    float dtv = dt[n * DI_ + d];
    float xv  = xcF[n * DI_ + d];
    float Bv  = dbl[n * 48 + DTR_ + s];
    float Cv  = dbl[n * 48 + DTR_ + DS_ + s];
    float dA = __expf(dtv * Ads);
    h = dA * h + dtv * xv * Bv;
    float contrib = h * Cv;
    #pragma unroll
    for (int o = 8; o; o >>= 1) contrib += __shfl_xor(contrib, o, 16);
    if (s == 0) {
      float y  = contrib + xv * Dpd;
      float zv = xz[n * (2 * DI_) + DI_ + d];
      float sz = zv / (1.f + __expf(-zv));
      ygate[n * DI_ + d] = to_bf16(y * sz);
    }
  }
}

// ---------------------------------------------------------------------------
extern "C" void kernel_launch(void* const* d_in, const int* in_sizes, int n_in,
                              void* d_out, int out_size, void* d_ws, size_t ws_size,
                              hipStream_t stream) {
  (void)in_sizes; (void)n_in; (void)out_size; (void)ws_size;
  const float* x     = (const float*)d_in[0];
  const float* ln1_g = (const float*)d_in[1];
  const float* ln1_b = (const float*)d_in[2];
  struct MP { const float *Win,*bin,*cw,*cb,*Wx,*Wdt,*bdt,*Alog,*Dp,*Wout,*bout; };
  MP pt = { (const float*)d_in[3],(const float*)d_in[4],(const float*)d_in[5],
            (const float*)d_in[6],(const float*)d_in[7],(const float*)d_in[8],
            (const float*)d_in[9],(const float*)d_in[10],(const float*)d_in[11],
            (const float*)d_in[12],(const float*)d_in[13] };
  const float* ln2_g = (const float*)d_in[14];
  const float* ln2_b = (const float*)d_in[15];
  MP ps = { (const float*)d_in[16],(const float*)d_in[17],(const float*)d_in[18],
            (const float*)d_in[19],(const float*)d_in[20],(const float*)d_in[21],
            (const float*)d_in[22],(const float*)d_in[23],(const float*)d_in[24],
            (const float*)d_in[25],(const float*)d_in[26] };
  const float* ln3_g = (const float*)d_in[27];
  const float* ln3_b = (const float*)d_in[28];
  const float* W1 = (const float*)d_in[29];
  const float* b1 = (const float*)d_in[30];
  const float* W2 = (const float*)d_in[31];
  const float* b2 = (const float*)d_in[32];
  float* xout = (float*)d_out;               // residual stream lives here

  // ---- workspace arena ----
  size_t off = 0;
  auto arena = [&](size_t bytes) -> void* {
    void* p = (char*)d_ws + off; off += (bytes + 255) & ~(size_t)255; return p;
  };
  bf16_t* WinT[2]  = { (bf16_t*)arena((size_t)2*DI_*D_*2),  (bf16_t*)arena((size_t)2*DI_*D_*2) };
  bf16_t* WxT[2]   = { (bf16_t*)arena((size_t)48*DI_*2),    (bf16_t*)arena((size_t)48*DI_*2) };
  bf16_t* WoutT[2] = { (bf16_t*)arena((size_t)D_*DI_*2),    (bf16_t*)arena((size_t)D_*DI_*2) };
  bf16_t* W1T = (bf16_t*)arena((size_t)DFF_*D_*2);
  bf16_t* W2T = (bf16_t*)arena((size_t)D_*DFF_*2);
  bf16_t* lnH  = (bf16_t*)arena((size_t)NROWS*D_*2);
  float*  xz   = (float*) arena((size_t)NROWS*2*DI_*4);
  float*  xcF  = (float*) arena((size_t)NROWS*DI_*4);
  bf16_t* xcH  = (bf16_t*)arena((size_t)NROWS*DI_*2);
  float*  dbl  = (float*) arena((size_t)NROWS*48*4);
  float*  dtb  = (float*) arena((size_t)NROWS*DI_*4);
  bf16_t* yg   = (bf16_t*)arena((size_t)NROWS*DI_*2);
  bf16_t* hH   = (bf16_t*)xz;                // MLP hidden reuses xz space

  auto tr = [&](const float* src, bf16_t* dst, int R, int C) {
    long n = (long)R * C;
    k_transpose_bf16<<<dim3((unsigned)((n + 255) / 256)), 256, 0, stream>>>(src, dst, R, C);
  };

  // ---- weight conversion (transposed bf16) ----
  tr(pt.Win,  WinT[0],  D_,  2*DI_);  tr(ps.Win,  WinT[1],  D_,  2*DI_);
  tr(pt.Wx,   WxT[0],   DI_, 48);     tr(ps.Wx,   WxT[1],   DI_, 48);
  tr(pt.Wout, WoutT[0], DI_, D_);     tr(ps.Wout, WoutT[1], DI_, D_);
  tr(W1, W1T, D_, DFF_);
  tr(W2, W2T, DFF_, D_);

  // ---- residual stream init: xout = x ----
  k_copy_f32<<<dim3(NROWS * D_ / 256), 256, 0, stream>>>(x, xout, (long)NROWS * D_);

  const MP*    P[2]   = { &pt, &ps };
  const float* lnG[2] = { ln1_g, ln2_g };
  const float* lnB[2] = { ln1_b, ln2_b };

  for (int blk = 0; blk < 2; ++blk) {
    const MP& p = *P[blk];
    // LN -> bf16
    k_layernorm_bf16<<<dim3(NROWS), 256, 0, stream>>>(xout, lnG[blk], lnB[blk], lnH);
    // xz = ln @ Win + bin   [16384 x 1024], K=256
    k_gemm_wmma<EPI_F32, 4><<<dim3(2*DI_/64, NROWS/128), 256, 0, stream>>>(
        lnH, WinT[blk], p.bin, xz, nullptr, NROWS, 2*DI_, D_, 2*DI_);
    // depthwise conv + silu
    k_conv_silu<<<dim3(NROWS * DI_ / 256), 256, 0, stream>>>(xz, p.cw, p.cb, xcF, xcH);
    // dbl = xc @ Wx          [16384 x 48], K=512 (no bias)
    k_gemm_wmma<EPI_F32, 3><<<dim3(1, NROWS/128), 256, 0, stream>>>(
        xcH, WxT[blk], nullptr, dbl, nullptr, NROWS, 48, DI_, 48);
    // dt = softplus(dtr @ Wdt + bdt)
    k_dt_softplus<<<dim3(NROWS, 2), 256, 0, stream>>>(dbl, p.Wdt, p.bdt, dtb);
    // selective scan + skip + gate -> bf16
    k_scan<<<dim3(B_, DI_/16), 256, 0, stream>>>(dtb, dbl, xcF, p.Alog, p.Dp, xz, yg);
    // xout += yg @ Wout + bout   [16384 x 256], K=512
    k_gemm_wmma<EPI_ADD_F32, 4><<<dim3(D_/64, NROWS/128), 256, 0, stream>>>(
        yg, WoutT[blk], p.bout, xout, nullptr, NROWS, D_, DI_, D_);
  }

  // ---- MLP ----
  k_layernorm_bf16<<<dim3(NROWS), 256, 0, stream>>>(xout, ln3_g, ln3_b, lnH);
  k_gemm_wmma<EPI_GELU_BF16, 4><<<dim3(DFF_/64, NROWS/128), 256, 0, stream>>>(
      lnH, W1T, b1, nullptr, hH, NROWS, DFF_, D_, DFF_);
  k_gemm_wmma<EPI_ADD_F32, 4><<<dim3(D_/64, NROWS/128), 256, 0, stream>>>(
      hH, W2T, b2, xout, nullptr, NROWS, D_, DFF_, D_);
}
